// QuantizedE8P12Linear_26912265077061
// MI455X (gfx1250) — matrix operand order repaired and checked by reading, not compiled, not run on hardware
//
#include <hip/hip_runtime.h>
#include <hip/hip_bf16.h>
#include <stdint.h>

typedef __attribute__((ext_vector_type(16))) _Float16 v16h;
typedef __attribute__((ext_vector_type(8)))  _Float16 v8h;
typedef __attribute__((ext_vector_type(8)))  float    v8f;
typedef int v4i __attribute__((__vector_size__(16)));   // matches builtin proto

#define AS1 __attribute__((address_space(1)))
#define AS3 __attribute__((address_space(3)))

#define TOKENS 2048
#define IN_F   4096   // K
#define M_OUT  4096   // N
#define NCODES 512

// Async global->LDS (CDNA5) if the toolchain exposes it; else reg staging.
#if defined(__gfx1250__) && __has_builtin(__builtin_amdgcn_global_load_async_to_lds_b128)
#define HAVE_ASYNC 1
#else
#define HAVE_ASYNC 0
#endif

// ---------------------------------------------------------------------------
// Kernel 1: decompress E8P codes -> fp16 weight rows (one code -> 8 halves)
// ---------------------------------------------------------------------------
__global__ __launch_bounds__(256) void decode_kernel(
    const int* __restrict__ Qidxs,
    const _Float16* __restrict__ grid_abs,       // [256][8] fp16
    const uint8_t* __restrict__ grid_abs_even,   // [256] bool
    _Float16* __restrict__ Wh)                   // [M_OUT][IN_F]
{
    int idx = blockIdx.x * 256 + threadIdx.x;    // 0 .. M_OUT*NCODES-1
    int code = Qidxs[idx];
    int abs_idx   = (code >> 8) & 255;
    int sign_bits = (code >> 1) & 127;
    int shift_bit =  code       & 1;
    int parity  = __popc(sign_bits) & 1;
    int abs_odd = 1 - (int)grid_abs_even[abs_idx];
    int sign0   = parity ^ abs_odd;

    v8h g = *(const v8h*)(grid_abs + abs_idx * 8);
    _Float16 shift = shift_bit ? (_Float16)0.25f : (_Float16)(-0.25f);

    v8h w;
    w[0] = (sign0 ? (_Float16)(-g[0]) : g[0]) + shift;
#pragma unroll
    for (int j = 1; j < 8; ++j) {
        int nb = (sign_bits >> (j - 1)) & 1;
        w[j] = (nb ? (_Float16)(-g[j]) : g[j]) + shift;
    }
    *(v8h*)(Wh + (size_t)idx * 8) = w;
}

// ---------------------------------------------------------------------------
// Kernel 2: x = FWHT_4096(input * SU) / 64  -> fp16   (one token per block)
// ---------------------------------------------------------------------------
__global__ __launch_bounds__(256) void fwht_in_kernel(
    const float* __restrict__ in, const float* __restrict__ SU,
    _Float16* __restrict__ Xh)
{
    __shared__ float buf[IN_F];
    const size_t t = blockIdx.x;
    const float* row = in + t * IN_F;
    for (int i = threadIdx.x; i < IN_F; i += 256) buf[i] = row[i] * SU[i];
    __syncthreads();
#pragma unroll 1
    for (int sh = 0; sh < 12; ++sh) {
#pragma unroll
        for (int p = 0; p < 8; ++p) {
            int k = threadIdx.x + p * 256;                       // 0..2047 pairs
            int i = ((k >> sh) << (sh + 1)) | (k & ((1 << sh) - 1));
            int s = 1 << sh;
            float a = buf[i], b = buf[i + s];
            buf[i] = a + b; buf[i + s] = a - b;
        }
        __syncthreads();
    }
    _Float16* orow = Xh + t * IN_F;
    for (int i = threadIdx.x; i < IN_F; i += 256)
        orow[i] = (_Float16)(buf[i] * 0.015625f);                // 1/sqrt(4096)
}

// ---------------------------------------------------------------------------
// Kernel 3: C (TOKENS x M_OUT, f32) = Xh (TOKENS x K) * Wh^T (M_OUT x K)
// Block tile 128x256, 8 waves (2x4), wave tile 64x64 (16 accumulators),
// K-tile 32 (= one WMMA K-step), double-buffered LDS, padded stride 40 halves.
// Per wave per k-tile: 16 ds_load_b128 -> 16 v_wmma; A-frags software-
// pipelined so LDS latency overlaps the previous group's WMMAs.
// Tile staging uses GLOBAL_LOAD_ASYNC_TO_LDS_B128 + s_wait_asynccnt.
// ---------------------------------------------------------------------------
#define TM  128
#define TN  256
#define TK  32
#define LDP 40                 // padded LDS row stride in halves (80 B)
#define KTILES (IN_F / TK)     // 128

__global__ __launch_bounds__(256) void gemm_kernel(
    const _Float16* __restrict__ A,   // Xh [TOKENS][IN_F]
    const _Float16* __restrict__ B,   // Wh [M_OUT][IN_F]
    float* __restrict__ C)            // [TOKENS][M_OUT]
{
    __shared__ _Float16 sA[2][TM * LDP];   // 2 x 10240 B
    __shared__ _Float16 sB[2][TN * LDP];   // 2 x 20480 B

    const int tid  = threadIdx.x;
    const int lane = tid & 31;
    const int wave = tid >> 5;
    const int wm   = wave >> 2;           // 0..1  (M)
    const int wn   = wave & 3;            // 0..3  (N)
    const int bm   = blockIdx.y * TM;
    const int bn   = blockIdx.x * TN;

    // staging map: 16B chunks; A has 512 chunks (2/thread), B 1024 (4/thread)
    const int r0 = tid >> 2;              // 0..63
    const int c0 = (tid & 3) * 8;         // half offset 0,8,16,24
    const _Float16* Ag0 = A + (size_t)(bm + r0) * IN_F + c0;
    const _Float16* Ag1 = Ag0 + (size_t)64 * IN_F;
    const _Float16* Bg0 = B + (size_t)(bn + r0) * IN_F + c0;
    const _Float16* Bg1 = Bg0 + (size_t)64  * IN_F;
    const _Float16* Bg2 = Bg0 + (size_t)128 * IN_F;
    const _Float16* Bg3 = Bg0 + (size_t)192 * IN_F;
    const int offA0 = r0 * LDP + c0;
    const int offA1 = (r0 + 64) * LDP + c0;
    const int offB0 = r0 * LDP + c0;
    const int offB1 = (r0 + 64)  * LDP + c0;
    const int offB2 = (r0 + 128) * LDP + c0;
    const int offB3 = (r0 + 192) * LDP + c0;

    v8f acc[4][4] = {};

#if HAVE_ASYNC
#define ASYNC_CP(gp, lp)                                                       \
    __builtin_amdgcn_global_load_async_to_lds_b128(                            \
        (AS1 v4i*)(gp), (AS3 v4i*)(lp), 0, 0)

#define TILE_FETCH(buf, koff)                                                  \
    do {                                                                       \
        ASYNC_CP(Ag0 + (koff), &sA[buf][offA0]);                               \
        ASYNC_CP(Ag1 + (koff), &sA[buf][offA1]);                               \
        ASYNC_CP(Bg0 + (koff), &sB[buf][offB0]);                               \
        ASYNC_CP(Bg1 + (koff), &sB[buf][offB1]);                               \
        ASYNC_CP(Bg2 + (koff), &sB[buf][offB2]);                               \
        ASYNC_CP(Bg3 + (koff), &sB[buf][offB3]);                               \
    } while (0)

#if __has_builtin(__builtin_amdgcn_s_wait_asynccnt)
#define ASYNC_WAIT() __builtin_amdgcn_s_wait_asynccnt(0)
#else
#define ASYNC_WAIT() asm volatile("s_wait_asynccnt 0x0" ::: "memory")
#endif

    TILE_FETCH(0, 0);
    ASYNC_WAIT();
    __syncthreads();
#else
    {
        uint4 a0 = *(const uint4*)Ag0, a1 = *(const uint4*)Ag1;
        uint4 b0 = *(const uint4*)Bg0, b1 = *(const uint4*)Bg1;
        uint4 b2 = *(const uint4*)Bg2, b3 = *(const uint4*)Bg3;
        *(uint4*)&sA[0][offA0] = a0; *(uint4*)&sA[0][offA1] = a1;
        *(uint4*)&sB[0][offB0] = b0; *(uint4*)&sB[0][offB1] = b1;
        *(uint4*)&sB[0][offB2] = b2; *(uint4*)&sB[0][offB3] = b3;
    }
    __syncthreads();
#endif

    const int ar = lane & 15;
    const int ak = (lane < 16) ? 0 : 16;   // A frag K base (contiguous k = 16h + s, same order as B)
    const int bk = (lane < 16) ? 0 : 16;   // B frag K base

#pragma unroll 1
    for (int kt = 0; kt < KTILES; ++kt) {
        const int cur = kt & 1;
        const int nxt = cur ^ 1;

#if HAVE_ASYNC
        if (kt < KTILES - 1) TILE_FETCH(nxt, (kt + 1) * TK);
#else
        uint4 na0, na1, nb0, nb1, nb2, nb3;
        if (kt < KTILES - 1) {
            const int koff = (kt + 1) * TK;
            na0 = *(const uint4*)(Ag0 + koff); na1 = *(const uint4*)(Ag1 + koff);
            nb0 = *(const uint4*)(Bg0 + koff); nb1 = *(const uint4*)(Bg1 + koff);
            nb2 = *(const uint4*)(Bg2 + koff); nb3 = *(const uint4*)(Bg3 + koff);
        }
#endif
        if (kt < KTILES - 2) {
            __builtin_prefetch(Ag0 + (kt + 2) * TK, 0, 3);
            __builtin_prefetch(Bg0 + (kt + 2) * TK, 0, 3);
        }

        const _Float16* la = &sA[cur][0];
        const _Float16* lb = &sB[cur][0];

        // B fragments held for the whole k-tile
        v16h bfrag[4];
#pragma unroll
        for (int j = 0; j < 4; ++j) {
            const _Float16* p = lb + (wn * 64 + j * 16 + ar) * LDP + bk;
            v8h lo = *(const v8h*)p;          // K = bk .. bk+7
            v8h hi = *(const v8h*)(p + 8);    // K = bk+8 .. bk+15
            bfrag[j] = __builtin_shufflevector(lo, hi,
                0,1,2,3,4,5,6,7,8,9,10,11,12,13,14,15);
        }

        // A fragments software-pipelined: load i+1 before group i's WMMAs
        auto loadA = [&](int i) -> v16h {
            const _Float16* p = la + (wm * 64 + i * 16 + ar) * LDP + ak;
            v8h lo = *(const v8h*)p;          // K = ak .. ak+7
            v8h hi = *(const v8h*)(p + 8);    // K = ak+8 .. ak+15
            return __builtin_shufflevector(lo, hi,
                0,1,2,3,4,5,6,7,8,9,10,11,12,13,14,15);
        };

        v16h aCur = loadA(0);
#pragma unroll
        for (int i = 0; i < 4; ++i) {
            v16h aNxt;
            if (i < 3) aNxt = loadA(i + 1);
#pragma unroll
            for (int j = 0; j < 4; ++j)
                acc[i][j] = __builtin_amdgcn_wmma_f32_16x16x32_f16(
                    false, aCur, false, bfrag[j],
                    (short)0, acc[i][j], false, false);
            if (i < 3) aCur = aNxt;
        }

#if HAVE_ASYNC
        if (kt < KTILES - 1) ASYNC_WAIT();
        __syncthreads();
#else
        if (kt < KTILES - 1) {
            *(uint4*)&sA[nxt][offA0] = na0; *(uint4*)&sA[nxt][offA1] = na1;
            *(uint4*)&sB[nxt][offB0] = nb0; *(uint4*)&sB[nxt][offB1] = nb1;
            *(uint4*)&sB[nxt][offB2] = nb2; *(uint4*)&sB[nxt][offB3] = nb3;
        }
        __syncthreads();
#endif
    }

    // epilogue: C/D layout -> VGPR v: lanes 0-15 row v, lanes 16-31 row v+8
    const int rowAdd = (lane < 16) ? 0 : 8;
#pragma unroll
    for (int i = 0; i < 4; ++i)
#pragma unroll
        for (int j = 0; j < 4; ++j) {
            const int row0 = bm + wm * 64 + i * 16 + rowAdd;
            const int col  = bn + wn * 64 + j * 16 + ar;
#pragma unroll
            for (int v = 0; v < 8; ++v)
                C[(size_t)(row0 + v) * M_OUT + col] = acc[i][j][v];
        }
}

// ---------------------------------------------------------------------------
// Kernel 4: in-place per-token FWHT on output, * Wscale/64 * SV
// ---------------------------------------------------------------------------
__global__ __launch_bounds__(256) void fwht_out_kernel(
    float* __restrict__ y, const float* __restrict__ SV,
    const float* __restrict__ Wscale)
{
    __shared__ float buf[M_OUT];
    const size_t t = blockIdx.x;
    float* row = y + t * M_OUT;
    for (int i = threadIdx.x; i < M_OUT; i += 256) buf[i] = row[i];
    __syncthreads();
#pragma unroll 1
    for (int sh = 0; sh < 12; ++sh) {
#pragma unroll
        for (int p = 0; p < 8; ++p) {
            int k = threadIdx.x + p * 256;
            int i = ((k >> sh) << (sh + 1)) | (k & ((1 << sh) - 1));
            int s = 1 << sh;
            float a = buf[i], b = buf[i + s];
            buf[i] = a + b; buf[i + s] = a - b;
        }
        __syncthreads();
    }
    const float sc = Wscale[0] * 0.015625f;
    for (int i = threadIdx.x; i < M_OUT; i += 256)
        row[i] = buf[i] * sc * SV[i];
}

// ---------------------------------------------------------------------------
// Launch
// ---------------------------------------------------------------------------
extern "C" void kernel_launch(void* const* d_in, const int* in_sizes, int n_in,
                              void* d_out, int out_size, void* d_ws, size_t ws_size,
                              hipStream_t stream)
{
    const float*    input    = (const float*)d_in[0];     // [2048][4096]
    const int*      Qidxs    = (const int*)d_in[1];       // [4096][512]
    const float*    SU       = (const float*)d_in[2];     // [4096]
    const float*    SV       = (const float*)d_in[3];     // [4096]
    const float*    Wscale   = (const float*)d_in[4];     // [1]
    const _Float16* grid_abs = (const _Float16*)d_in[9];  // [256][8] fp16
    const uint8_t*  grid_evn = (const uint8_t*)d_in[10];  // [256] bool

    // workspace: Xh (16 MB) then Wh (32 MB)
    _Float16* Xh = (_Float16*)d_ws;
    _Float16* Wh = Xh + (size_t)TOKENS * IN_F;
    float*    C  = (float*)d_out;

    decode_kernel<<<(M_OUT * NCODES) / 256, 256, 0, stream>>>(
        Qidxs, grid_abs, grid_evn, Wh);

    fwht_in_kernel<<<TOKENS, 256, 0, stream>>>(input, SU, Xh);

    dim3 grid(M_OUT / TN, TOKENS / TM);   // (16, 16)
    gemm_kernel<<<grid, 256, 0, stream>>>(Xh, Wh, C);

    fwht_out_kernel<<<TOKENS, 256, 0, stream>>>(C, SV, Wscale);
}
